// BERT_LSTM_CRF_18184891531320
// MI455X (gfx1250) — matrix operand order repaired and checked by reading, not compiled, not run on hardware
//
#include <hip/hip_runtime.h>
#include <stdint.h>

// Problem constants (from reference)
#define B_    32
#define S_    512
#define V_    100000
#define DE_   256
#define NE_   3
#define EMB_  512
#define HID_  512
#define TAG_  22
#define G4_   2048           // 4*HID
#define KIN_  768            // NE*DE
#define M_    (B_ * S_)      // 16384 rows

typedef __bf16 bf16_t;
typedef bf16_t v16bf __attribute__((ext_vector_type(16)));
typedef float  v8f   __attribute__((ext_vector_type(8)));
typedef unsigned short u16;
typedef unsigned int   u32;

#ifndef __has_builtin
#define __has_builtin(x) 0
#endif
#if __has_builtin(__builtin_amdgcn_tensor_load_to_lds) && \
    __has_builtin(__builtin_amdgcn_s_wait_tensorcnt)
#define HAVE_TDM 1
#else
#define HAVE_TDM 0
#endif

union FragBF { v16bf v; uint4 q[2]; };

__device__ __forceinline__ u16 f2bf(float f) {
  u32 u = __float_as_uint(f);
  u32 r = (u + 0x7FFFu + ((u >> 16) & 1u)) >> 16;   // round-to-nearest-even
  return (u16)r;
}
__device__ __forceinline__ float bf2f(u16 b) {
  return __uint_as_float((u32)b << 16);
}
__device__ __forceinline__ float sigmf(float x) { return 1.0f / (1.0f + __expf(-x)); }

__device__ __forceinline__ v8f vzero8() {
  v8f z;
#pragma unroll
  for (int i = 0; i < 8; ++i) z[i] = 0.0f;
  return z;
}

// ---- WMMA fragment helpers (CDNA5 §7.12.2 layouts, wave32) ----------------
// A: 16x32 bf16 tile from row-major (M x K). Lane l holds row (l&15);
// elements 0..7 = K kbase+8*(l>>4)..+7, elements 8..15 = K kbase+16+8*(l>>4)..+7.
__device__ __forceinline__ v16bf load_a_frag(const u16* __restrict__ A, int lda,
                                             int mbase, int kbase, int lane) {
  FragBF f;
  int half = lane >> 4;
  int m    = lane & 15;
  const u16* p = A + (size_t)(mbase + m) * lda + kbase + 8 * half;
  f.q[0] = *reinterpret_cast<const uint4*>(p);
  f.q[1] = *reinterpret_cast<const uint4*>(p + 16);
  return f.v;
}

// B: 32x16 (K x N) bf16 tile. Weights stored "B-ready": row-major (N x K), so
// lane l (column n = nbase + (l&15)) reads elements 0..7 = K kbase+8*(l>>4)..+7,
// elements 8..15 = K kbase+16+8*(l>>4)..+7.
__device__ __forceinline__ v16bf load_b_frag(const u16* __restrict__ W, int ldk,
                                             int nbase, int kbase, int lane) {
  FragBF f;
  int half = lane >> 4;
  int n    = lane & 15;
  const u16* p = W + (size_t)(nbase + n) * ldk + kbase + 8 * half;
  f.q[0] = *reinterpret_cast<const uint4*>(p);
  f.q[1] = *reinterpret_cast<const uint4*>(p + 16);
  return f.v;
}

__device__ __forceinline__ v8f wmma_bf16(v16bf a, v16bf b, v8f c) {
  return __builtin_amdgcn_wmma_f32_16x16x32_bf16(false, a, false, b,
                                                 (short)0, c, false, false);
}

#if HAVE_TDM
typedef unsigned int u32x4 __attribute__((ext_vector_type(4)));
typedef int          i32x4 __attribute__((ext_vector_type(4)));
typedef int          i32x8 __attribute__((ext_vector_type(8)));

// TDM: DMA one contiguous 128 KB xg step-slab (32768 dwords) from global into
// LDS. D# per ISA §8.3/8.4: 1-D dword tile, tensor_dim0 = tile_dim0 = 32768.
// This toolchain exposes the 6-operand builtin:
//   (u32x4 g0, i32x8 g1, i32x4 g2, i32x4 g3, i32x8 extra, int cpol)
__device__ __forceinline__ void tdm_load_slab(const u16* gsrc, u32 lds_byte_off) {
  unsigned long long ga = (unsigned long long)(size_t)gsrc;
  u32x4 g0;
  g0.x = 1u;                                            // count=1 user D#
  g0.y = lds_byte_off;                                  // lds_addr [63:32]
  g0.z = (u32)ga;                                       // global_addr[31:0]
  g0.w = (u32)((ga >> 32) & 0x1FFFFFFu) | (2u << 30);   // ga[56:32] | type=2
  i32x8 g1;
  g1[0] = (int)(2u << 16);        // workgroup_mask=0, data_size=4B, no flags
  g1[1] = (int)(0x8000u << 16);   // tensor_dim0[15:0]=32768 in bits[31:16]
  g1[2] = (int)(1u << 16);        // tensor_dim0[31:16]=0 | tensor_dim1=1
  g1[3] = (int)(0x8000u << 16);   // tensor_dim1 hi=0 | tile_dim0=32768
  g1[4] = 0;                      // tile_dim1=0 (1-D), tile_dim2=0
  g1[5] = 32768;                  // tensor_dim0_stride[31:0]
  g1[6] = 0;
  g1[7] = 0;
  i32x4 g2; g2[0] = 0; g2[1] = 0; g2[2] = 0; g2[3] = 0;
  i32x4 g3; g3[0] = 0; g3[1] = 0; g3[2] = 0; g3[3] = 0;
  i32x8 g4;
#pragma unroll
  for (int i = 0; i < 8; ++i) g4[i] = 0;
  __builtin_amdgcn_tensor_load_to_lds(g0, g1, g2, g3, g4, 0);
}
#endif

// ---------------------------------------------------------------------------
// Tiny prep kernels
// ---------------------------------------------------------------------------
__global__ void softmax3_kernel(const float* __restrict__ a, float* __restrict__ w) {
  float m  = fmaxf(a[0], fmaxf(a[1], a[2]));
  float e0 = __expf(a[0] - m), e1 = __expf(a[1] - m), e2 = __expf(a[2] - m);
  float s  = e0 + e1 + e2;
  w[0] = e0 / s; w[1] = e1 / s; w[2] = e2 / s;
}

__global__ void __launch_bounds__(256)
conv_bf16_kernel(const float* __restrict__ src, u16* __restrict__ dst, int n) {
  int i = blockIdx.x * 256 + threadIdx.x;
  if (i < n) dst[i] = f2bf(src[i]);
}

// W1 (KIN_ x EMB_) -> W1T (EMB_ x KIN_) bf16  (B-ready layout)
__global__ void __launch_bounds__(256)
conv_w1t_kernel(const float* __restrict__ w1, u16* __restrict__ w1t) {
  int i = blockIdx.x * 256 + threadIdx.x;
  if (i < EMB_ * KIN_) {
    int n = i / KIN_, k = i - n * KIN_;
    w1t[i] = f2bf(w1[(size_t)k * EMB_ + n]);
  }
}

// Wout (1024 x 22) -> WoutT (32 x 1024) bf16, zero padded N=22..31
__global__ void __launch_bounds__(256)
conv_woutt_kernel(const float* __restrict__ wo, u16* __restrict__ wt) {
  int i = blockIdx.x * 256 + threadIdx.x;
  if (i < 32 * 1024) {
    int n = i >> 10, k = i & 1023;
    float v = (n < TAG_) ? wo[(size_t)k * TAG_ + n] : 0.0f;
    wt[i] = f2bf(v);
  }
}

__global__ void __launch_bounds__(256)
bias_sum_kernel(const float* __restrict__ a, const float* __restrict__ b,
                float* __restrict__ o, int n) {
  int i = blockIdx.x * 256 + threadIdx.x;
  if (i < n) o[i] = a[i] + b[i];
}

// Weighted embedding gather: Xin[bs][ne*256+d] = bf16(w[ne]*tab[ne][tok][d])
__global__ void __launch_bounds__(256)
gather_embed_kernel(const int* __restrict__ tok, const float* __restrict__ tabs,
                    const float* __restrict__ w3, u16* __restrict__ xin) {
  int bs = blockIdx.x;
  int t  = tok[bs];
  int d  = threadIdx.x;
#pragma unroll
  for (int ne = 0; ne < NE_; ++ne) {
    float v = tabs[((size_t)ne * V_ + (size_t)t) * DE_ + d] * w3[ne];
    xin[(size_t)bs * KIN_ + ne * DE_ + d] = f2bf(v);
  }
}

// ---------------------------------------------------------------------------
// Generic WMMA GEMM: C[M x N] = A[M x K](bf16) * W[N x K]^T(bf16) + bias[N]
// Each wave computes a 32(M) x 64(N) tile (8 fp32 accumulators).
// Block = 8 waves covering 256 M rows x 64 N cols.
// MODE 0: output bf16, row-major (row*N + n)            (X projection)
// MODE 1: output bf16, t-major xg layout ((s*B + b)*G4_ + n), row = b*S_+s
// ---------------------------------------------------------------------------
template <int MODE>
__global__ void __launch_bounds__(256)
gemm_wmma_kernel(const u16* __restrict__ A, const u16* __restrict__ W,
                 const float* __restrict__ bias, u16* __restrict__ out,
                 int K, int N) {
  int lane = threadIdx.x & 31;
  int w    = threadIdx.x >> 5;
  int m0   = blockIdx.y * 256 + w * 32;
  int n0   = blockIdx.x * 64;

  v8f acc[2][4];
#pragma unroll
  for (int mi = 0; mi < 2; ++mi)
#pragma unroll
    for (int t = 0; t < 4; ++t) acc[mi][t] = vzero8();

  for (int kb = 0; kb < K; kb += 32) {
    v16bf a0 = load_a_frag(A, K, m0,      kb, lane);
    v16bf a1 = load_a_frag(A, K, m0 + 16, kb, lane);
#pragma unroll
    for (int t = 0; t < 4; ++t) {
      v16bf b = load_b_frag(W, K, n0 + 16 * t, kb, lane);
      acc[0][t] = wmma_bf16(a0, b, acc[0][t]);
      acc[1][t] = wmma_bf16(a1, b, acc[1][t]);
    }
  }

  int half = lane >> 4;
  int nl   = lane & 15;
#pragma unroll
  for (int mi = 0; mi < 2; ++mi) {
#pragma unroll
    for (int t = 0; t < 4; ++t) {
      int n    = n0 + 16 * t + nl;
      float bs = bias[n];
#pragma unroll
      for (int j = 0; j < 8; ++j) {
        int row = m0 + 16 * mi + j + 8 * half;
        float v = acc[mi][t][j] + bs;
        if (MODE == 0) {
          out[(size_t)row * N + n] = f2bf(v);
        } else {
          int b = row >> 9;            // row = b*S_ + s, S_ = 512
          int s = row & (S_ - 1);
          out[((size_t)s * B_ + b) * G4_ + n] = f2bf(v);
        }
      }
    }
  }
}

// ---------------------------------------------------------------------------
// Persistent bidirectional LSTM recurrence. grid = 2 (dir), block = 1024
// (32 waves). h lives in LDS (32x512 bf16 = 32 KB); c lives in registers.
// The next step's 128 KB xg gate slab is prefetched into a double-buffered
// LDS region by the Tensor Data Mover while this step's 128 WMMAs run
// (s_wait_tensorcnt + barrier at the step boundary). Total LDS = 288 KB.
// ---------------------------------------------------------------------------
__global__ void __launch_bounds__(1024)
lstm_persistent_kernel(const u16* __restrict__ xg_f, const u16* __restrict__ xg_r,
                       const u16* __restrict__ whh_f, const u16* __restrict__ whh_r,
                       u16* __restrict__ h_f, u16* __restrict__ h_r) {
  __shared__ __align__(16) u16 hsh[B_ * HID_];        // 32 KB: h_{t-1}
#if HAVE_TDM
  __shared__ __align__(16) u16 xgslab[2][B_ * G4_];   // 2 x 128 KB gate slabs
#endif

  int dir = blockIdx.x;
  const u16* xg   = dir ? xg_r : xg_f;    // (S, B, 4H) bf16, t-major
  const u16* whh  = dir ? whh_r : whh_f;  // (2048 x 512) B-ready bf16
  u16*       hout = dir ? h_r : h_f;      // (B, S, H) bf16

  int tid  = threadIdx.x;
  int lane = tid & 31;
  int w    = tid >> 5;                    // 0..31 : H-column tile
  int half = lane >> 4;
  int nl   = lane & 15;
  int ncol = w * 16 + nl;                 // this lane's H column

  for (int i = tid; i < B_ * HID_; i += 1024) hsh[i] = 0;

#if HAVE_TDM
  if (tid == 0) {                         // TDM ignores EXEC; one wave issues
    int t0 = dir ? (S_ - 1) : 0;
    tdm_load_slab(xg + (size_t)t0 * B_ * G4_, (u32)(size_t)&xgslab[0][0]);
    __builtin_amdgcn_s_wait_tensorcnt(0);
  }
#endif
  __syncthreads();

  float c[2][8];
#pragma unroll
  for (int mi = 0; mi < 2; ++mi)
#pragma unroll
    for (int j = 0; j < 8; ++j) c[mi][j] = 0.0f;

  for (int step = 0; step < S_; ++step) {
    int t   = dir ? (S_ - 1 - step) : step;
    int cur = step & 1;

#if HAVE_TDM
    // Prefetch next step's gate slab while this step's WMMAs run.
    if (tid == 0 && step + 1 < S_) {
      int tn = dir ? (S_ - 2 - step) : (step + 1);
      tdm_load_slab(xg + (size_t)tn * B_ * G4_,
                    (u32)(size_t)&xgslab[cur ^ 1][0]);
    }
#endif

    u16 hbits[2][8];
#pragma unroll
    for (int mi = 0; mi < 2; ++mi) {
      v8f ai = vzero8(), af = vzero8(), ag = vzero8(), ao = vzero8();
      int m = mi * 16 + nl;               // batch row for A fragment

      for (int kb = 0; kb < HID_; kb += 32) {
        FragBF fa;
        const u16* pa = &hsh[m * HID_ + kb + 8 * half];
        fa.q[0] = *reinterpret_cast<const uint4*>(pa);
        fa.q[1] = *reinterpret_cast<const uint4*>(pa + 16);
        v16bf a = fa.v;

        v16bf bi = load_b_frag(whh, HID_, 0 * HID_ + w * 16, kb, lane);
        v16bf bf = load_b_frag(whh, HID_, 1 * HID_ + w * 16, kb, lane);
        v16bf bg = load_b_frag(whh, HID_, 2 * HID_ + w * 16, kb, lane);
        v16bf bo = load_b_frag(whh, HID_, 3 * HID_ + w * 16, kb, lane);

        ai = wmma_bf16(a, bi, ai);
        af = wmma_bf16(a, bf, af);
        ag = wmma_bf16(a, bg, ag);
        ao = wmma_bf16(a, bo, ao);
      }

#pragma unroll
      for (int j = 0; j < 8; ++j) {
        int b = mi * 16 + j + 8 * half;
#if HAVE_TDM
        const u16* gsl = &xgslab[cur][b * G4_ + ncol];
        float gi = ai[j] + bf2f(gsl[0 * HID_]);
        float gf = af[j] + bf2f(gsl[1 * HID_]);
        float gg = ag[j] + bf2f(gsl[2 * HID_]);
        float go = ao[j] + bf2f(gsl[3 * HID_]);
#else
        size_t xb = ((size_t)t * B_ + b) * G4_ + ncol;
        float gi = ai[j] + bf2f(xg[xb + 0 * HID_]);
        float gf = af[j] + bf2f(xg[xb + 1 * HID_]);
        float gg = ag[j] + bf2f(xg[xb + 2 * HID_]);
        float go = ao[j] + bf2f(xg[xb + 3 * HID_]);
#endif
        float cc = c[mi][j];
        cc = sigmf(gf) * cc + sigmf(gi) * tanhf(gg);
        c[mi][j] = cc;
        float h = sigmf(go) * tanhf(cc);
        u16 hb = f2bf(h);
        hbits[mi][j] = hb;
        hout[(size_t)b * (S_ * HID_) + (size_t)t * HID_ + ncol] = hb;
      }
    }

    __syncthreads();   // all reads of h_{t-1} and xg slab complete
#pragma unroll
    for (int mi = 0; mi < 2; ++mi)
#pragma unroll
      for (int j = 0; j < 8; ++j) {
        int b = mi * 16 + j + 8 * half;
        hsh[b * HID_ + ncol] = hbits[mi][j];
      }
#if HAVE_TDM
    if (w == 0 && step + 1 < S_)          // issuing wave drains TENSORcnt
      __builtin_amdgcn_s_wait_tensorcnt(0);
#endif
    __syncthreads();   // h_t (and next slab) visible to all waves
  }
}

// ---------------------------------------------------------------------------
// Output projection: out[M x 22] = [Hf | Hr](M x 1024, bf16) * WoutT + bout
// Each wave: one 16-row M tile, both N tiles (n<22 guarded on store).
// ---------------------------------------------------------------------------
__global__ void __launch_bounds__(256)
out_gemm_kernel(const u16* __restrict__ Hf, const u16* __restrict__ Hr,
                const u16* __restrict__ Wt, const float* __restrict__ bout,
                float* __restrict__ out) {
  int lane = threadIdx.x & 31;
  int w    = threadIdx.x >> 5;
  int m0   = (blockIdx.x * 8 + w) * 16;

  v8f acc[2] = { vzero8(), vzero8() };

  for (int kb = 0; kb < 2 * HID_; kb += 32) {
    const u16* src = (kb < HID_) ? Hf : Hr;
    int kk = kb & (HID_ - 1);
    v16bf a = load_a_frag(src, HID_, m0, kk, lane);
#pragma unroll
    for (int t = 0; t < 2; ++t) {
      v16bf b = load_b_frag(Wt, 2 * HID_, 16 * t, kb, lane);
      acc[t] = wmma_bf16(a, b, acc[t]);
    }
  }

  int half = lane >> 4;
  int nl   = lane & 15;
#pragma unroll
  for (int t = 0; t < 2; ++t) {
    int n = 16 * t + nl;
    if (n < TAG_) {
      float bs = bout[n];
#pragma unroll
      for (int j = 0; j < 8; ++j) {
        int row = m0 + j + 8 * half;
        out[(size_t)row * TAG_ + n] = acc[t][j] + bs;
      }
    }
  }
}

// ---------------------------------------------------------------------------
extern "C" void kernel_launch(void* const* d_in, const int* in_sizes, int n_in,
                              void* d_out, int out_size, void* d_ws, size_t ws_size,
                              hipStream_t stream) {
  (void)in_sizes; (void)n_in; (void)out_size; (void)ws_size;

  const int*   tok   = (const int*)  d_in[0];
  const float* tabs  = (const float*)d_in[1];
  const float* arch  = (const float*)d_in[2];
  const float* W1    = (const float*)d_in[3];
  const float* b1    = (const float*)d_in[4];
  const float* Wih_f = (const float*)d_in[5];
  const float* Whh_f = (const float*)d_in[6];
  const float* bih_f = (const float*)d_in[7];
  const float* bhh_f = (const float*)d_in[8];
  const float* Wih_r = (const float*)d_in[9];
  const float* Whh_r = (const float*)d_in[10];
  const float* bih_r = (const float*)d_in[11];
  const float* bhh_r = (const float*)d_in[12];
  const float* Wout  = (const float*)d_in[13];
  const float* bout  = (const float*)d_in[14];
  float* out = (float*)d_out;

  // Workspace carve-out (256 B aligned)
  char* p = (char*)d_ws;
  auto carve = [&](size_t bytes) -> char* {
    char* r = p;
    p += (bytes + 255) & ~(size_t)255;
    return r;
  };
  float* w3   = (float*)carve(16);
  u16*   Xin  = (u16*)  carve((size_t)M_  * KIN_ * 2);
  u16*   W1T  = (u16*)  carve((size_t)EMB_* KIN_ * 2);
  u16*   Xbf  = (u16*)  carve((size_t)M_  * EMB_ * 2);
  u16*   WihF = (u16*)  carve((size_t)G4_ * EMB_ * 2);
  u16*   WihR = (u16*)  carve((size_t)G4_ * EMB_ * 2);
  u16*   WhhF = (u16*)  carve((size_t)G4_ * HID_ * 2);
  u16*   WhhR = (u16*)  carve((size_t)G4_ * HID_ * 2);
  u16*   WoT  = (u16*)  carve((size_t)32 * 1024 * 2);
  float* bsF  = (float*)carve((size_t)G4_ * 4);
  float* bsR  = (float*)carve((size_t)G4_ * 4);
  u16*   xgF  = (u16*)  carve((size_t)M_ * G4_ * 2);
  u16*   xgR  = (u16*)  carve((size_t)M_ * G4_ * 2);
  u16*   Hf   = (u16*)  carve((size_t)M_ * HID_ * 2);
  u16*   Hr   = (u16*)  carve((size_t)M_ * HID_ * 2);

  // 1) arch softmax
  softmax3_kernel<<<1, 1, 0, stream>>>(arch, w3);

  // 2) weight conversions (bf16, B-ready layouts) + bias sums
  conv_w1t_kernel<<<(EMB_ * KIN_ + 255) / 256, 256, 0, stream>>>(W1, W1T);
  conv_bf16_kernel<<<(G4_ * EMB_ + 255) / 256, 256, 0, stream>>>(Wih_f, WihF, G4_ * EMB_);
  conv_bf16_kernel<<<(G4_ * EMB_ + 255) / 256, 256, 0, stream>>>(Wih_r, WihR, G4_ * EMB_);
  conv_bf16_kernel<<<(G4_ * HID_ + 255) / 256, 256, 0, stream>>>(Whh_f, WhhF, G4_ * HID_);
  conv_bf16_kernel<<<(G4_ * HID_ + 255) / 256, 256, 0, stream>>>(Whh_r, WhhR, G4_ * HID_);
  conv_woutt_kernel<<<(32 * 1024 + 255) / 256, 256, 0, stream>>>(Wout, WoT);
  bias_sum_kernel<<<(G4_ + 255) / 256, 256, 0, stream>>>(bih_f, bhh_f, bsF, G4_);
  bias_sum_kernel<<<(G4_ + 255) / 256, 256, 0, stream>>>(bih_r, bhh_r, bsR, G4_);

  // 3) weighted embedding gather -> Xin (bf16, 16384 x 768)
  gather_embed_kernel<<<M_, DE_, 0, stream>>>(tok, tabs, w3, Xin);

  // 4) X = Xin @ W1 + b1 -> bf16 (16384 x 512)
  {
    dim3 g(EMB_ / 64, M_ / 256);
    gemm_wmma_kernel<0><<<g, 256, 0, stream>>>(Xin, W1T, b1, Xbf, KIN_, EMB_);
  }

  // 5) xg = X @ Wih^T + (bih+bhh) -> bf16, t-major, per direction
  {
    dim3 g(G4_ / 64, M_ / 256);
    gemm_wmma_kernel<1><<<g, 256, 0, stream>>>(Xbf, WihF, bsF, xgF, EMB_, G4_);
    gemm_wmma_kernel<1><<<g, 256, 0, stream>>>(Xbf, WihR, bsR, xgR, EMB_, G4_);
  }

  // 6) persistent bidirectional LSTM recurrence (1 workgroup per direction,
  //    TDM double-buffered gate-slab prefetch into LDS)
  lstm_persistent_kernel<<<2, 1024, 0, stream>>>(xgF, xgR, WhhF, WhhR, Hf, Hr);

  // 7) out = [Hf|Hr] @ Wout + bout
  out_gemm_kernel<<<M_ / 16 / 8, 256, 0, stream>>>(Hf, Hr, WoT, bout, out);
}
